// EncoderGIN_44908178047490
// MI455X (gfx1250) — matrix operand order repaired and checked by reading, not compiled, not run on hardware
//
#include <hip/hip_runtime.h>

#define N_NODES 100000
#define N_EDGES 1600000
#define DIM     128
#define LAYERS  3
#define BN_EPS  1e-5f
#define ROWS    64          // rows per MLP block (4 row-tiles of 16)

typedef __attribute__((ext_vector_type(16))) __bf16 v16bf;
typedef __attribute__((ext_vector_type(8)))  float  v8f;

// CDNA5 async Global->LDS path (guarded: falls back to VGPR copy if the
// toolchain doesn't expose the builtin).
#if defined(__has_builtin)
#if __has_builtin(__builtin_amdgcn_global_load_async_to_lds_b128)
#define HAVE_ASYNC_LDS 1
#endif
#endif

#if defined(HAVE_ASYNC_LDS)
// hipcc diagnostic shows the builtin takes:
//   arg0: __attribute__((vector_size(16))) int  AS(1)*   (global, non-const)
//   arg1: same vector type in AS(3)                      (LDS)
//   arg2, arg3: int immediates (offset, cpol)
typedef int v4i_vs __attribute__((vector_size(16)));
typedef __attribute__((address_space(1))) v4i_vs* gv4i_p;
typedef __attribute__((address_space(3))) v4i_vs* lv4i_p;
#endif

static __device__ __forceinline__ void wait_async0() {
#if defined(__has_builtin) && __has_builtin(__builtin_amdgcn_s_wait_asynccnt)
    __builtin_amdgcn_s_wait_asynccnt(0);
#else
    asm volatile("s_wait_asynccnt 0x0" ::: "memory");
#endif
}

// 32-byte fragment: loaded from LDS as two b128s, fed to WMMA as v16bf
union Frag {
    uint4          q[2];
    unsigned short u[16];
    v16bf          v;
};

// f32 -> bf16, round-to-nearest-even
static __device__ __forceinline__ unsigned short f2bf(float f) {
    unsigned int u = __float_as_uint(f);
    u += 0x7FFFu + ((u >> 16) & 1u);
    return (unsigned short)(u >> 16);
}

// ---------------------------------------------------------------------------
// Fragment-layout index helpers (ISA 7.12.2, 16-bit A 16x32 / B 32x16)
// ---------------------------------------------------------------------------
// B matrix (k,n) -> flat index in B-fragment-major layout
// [coltile c][kstep ks][lane][elem e], 16 contiguous elems per lane.
static __device__ __forceinline__ int swz_b(int k, int n) {
    int c   = n >> 4;
    int ks  = k >> 5;
    int rem = k & 31;
    int hib = (rem >> 3) & 1;     // lane half selects K 0..7,16..23 / 8..15,24..31
    int e   = (rem & 7) | ((rem >> 4) << 3);
    int lane = (hib << 4) | (n & 15);
    return (((c << 2) + ks) << 9) + (lane << 4) + e;
}

// A matrix: map k (0..127) to (ks, lane-half hi2, elem e) of the 16x32 A frag.
static __device__ __forceinline__ void swz_a_k(int k, int& ks, int& hi2, int& e) {
    ks      = k >> 5;
    int k32 = k & 31;
    int w   = k32 & 1;
    int q   = k32 >> 1;           // = base(p)/2 + hi*4
    hi2     = (q >> 2) & 1;
    int grp = q >> 3;
    int p   = (grp << 2) | (q & 3);
    e       = (p << 1) | w;
}

// ---------------------------------------------------------------------------
// Kernel 0: convert weights to bf16 AND pre-swizzle to B-fragment-major order.
// Wbf layout: [layer][0=W1,1=W2][16384] fragment-major.
// ---------------------------------------------------------------------------
__global__ void convw_kernel(const float* __restrict__ W1,
                             const float* __restrict__ W2,
                             unsigned short* __restrict__ Wbf) {
    int i = blockIdx.x * blockDim.x + threadIdx.x;
    const int per_layer = DIM * DIM;
    if (i < LAYERS * per_layer) {
        int l = i / per_layer;
        int r = i - l * per_layer;
        int k = r >> 7;           // W[k][n] row-major in the input
        int n = r & (DIM - 1);
        int dst = swz_b(k, n);
        Wbf[(l * 2 + 0) * per_layer + dst] = f2bf(W1[i]);
        Wbf[(l * 2 + 1) * per_layer + dst] = f2bf(W2[i]);
    }
}

// ---------------------------------------------------------------------------
// Kernel 1: agg = h   (GIN self term, eps = 0)
// ---------------------------------------------------------------------------
__global__ void init_agg_kernel(const float* __restrict__ h,
                                float* __restrict__ agg) {
    const long long total4 = (long long)N_NODES * DIM / 4;
    long long i = (long long)blockIdx.x * blockDim.x + threadIdx.x;
    long long stride = (long long)gridDim.x * blockDim.x;
    const float4* src = (const float4*)h;
    float4* dst = (float4*)agg;
    for (; i < total4; i += stride) dst[i] = src[i];
}

// ---------------------------------------------------------------------------
// Kernel 2: scatter-add over edge list. One wave32 per edge, 4 floats/lane.
// agg[dst] += h[src]; h (51.2 MB) is L2-resident on MI455X (192 MB L2).
// ---------------------------------------------------------------------------
__global__ void scatter_kernel(const float* __restrict__ h,
                               const long long* __restrict__ ei,
                               float* __restrict__ agg) {
    int edge = blockIdx.x * (blockDim.x >> 5) + (threadIdx.x >> 5);
    int lane = threadIdx.x & 31;
    if (edge >= N_EDGES) return;

    __builtin_prefetch(&ei[edge + 256], 0, 0);            // global_prefetch_b8
    __builtin_prefetch(&ei[N_EDGES + edge + 256], 0, 0);

    long long s = ei[edge];
    long long d = ei[N_EDGES + edge];

    const float4* hs = (const float4*)(h + s * DIM);
    float4 v = hs[lane];
    float* ad = agg + d * DIM + lane * 4;
    atomicAdd(ad + 0, v.x);
    atomicAdd(ad + 1, v.y);
    atomicAdd(ad + 2, v.z);
    atomicAdd(ad + 3, v.w);
}

// helper: stage one 32KB weight tile into LDS (async DMA if available)
static __device__ __forceinline__ void stage_w(const unsigned short* __restrict__ Wsrc,
                                               unsigned short* lds_w, int tid) {
#if defined(HAVE_ASYNC_LDS)
    for (int i = tid; i < DIM * DIM / 8; i += 256) {
        __builtin_amdgcn_global_load_async_to_lds_b128(
            (gv4i_p)(Wsrc + i * 8), (lv4i_p)(lds_w + i * 8), 0, 0);
    }
#else
    const uint4* src = (const uint4*)Wsrc;
    uint4* dst = (uint4*)lds_w;
    for (int i = tid; i < DIM * DIM / 8; i += 256) dst[i] = src[i];
#endif
}

// ---------------------------------------------------------------------------
// Kernel 3: fused  relu(agg@W1+b1) -> relu(@W2+b2) -> BN(eval) -> relu
// 256 threads = 8 waves; 64 rows x 128 cols per block.
// Wave w owns column tile w; 4 row-tiles each; B fragment reused per kstep.
// All fragments are contiguous 32B LDS reads (2x ds_load_b128).
// W tiles arrive via GLOBAL_LOAD_ASYNC_TO_LDS (ASYNCcnt), overlapped with
// the VALU staging/spill work.
// ---------------------------------------------------------------------------
__global__ void __launch_bounds__(256)
mlp_kernel(const float* __restrict__ agg,
           const unsigned short* __restrict__ W1bf,
           const float* __restrict__ b1,
           const unsigned short* __restrict__ W2bf,
           const float* __restrict__ b2,
           const float* __restrict__ gamma,
           const float* __restrict__ beta,
           const float* __restrict__ rmean,
           const float* __restrict__ rvar,
           float* __restrict__ out) {
    // W tile in B-fragment-major order (straight copy from pre-swizzled global)
    __shared__ __align__(16) unsigned short lds_w[DIM * DIM];    // 32 KB
    // activations in A-fragment-major order: [rowtile rt][ks][lane][16]
    // reused in-place for the MLP hidden between the two GEMMs
    __shared__ __align__(16) unsigned short lds_a[ROWS * DIM];   // 16 KB

    const int tid  = threadIdx.x;
    const int wave = tid >> 5;            // column tile 0..7
    const int lane = tid & 31;
    const int hi   = lane >> 4;
    const int lr   = lane & 15;
    const int r0   = blockIdx.x * ROWS;
    const int cn   = (wave << 4) + lr;    // this lane's output column (C layout)

    // --- kick off W1 DMA first; it streams while we do the A staging ---
    stage_w(W1bf, lds_w, tid);

    // --- stage activations (f32 -> bf16, permuted to A-fragment order) ---
    for (int i = tid; i < ROWS * DIM; i += 256) {
        int m = i >> 7;                   // 0..63
        int k = i & (DIM - 1);
        int rg = r0 + m; if (rg >= N_NODES) rg = N_NODES - 1;   // clamp tail
        int ks, hi2, e;
        swz_a_k(k, ks, hi2, e);
        int lane2 = (hi2 << 4) | (m & 15);
        int dst = ((((m >> 4) << 2) + ks) << 9) + (lane2 << 4) + e;
        lds_a[dst] = f2bf(agg[(long long)rg * DIM + k]);
    }
#if defined(HAVE_ASYNC_LDS)
    wait_async0();
#endif
    __syncthreads();

    // ===================== GEMM 1: mid = relu(agg @ W1 + b1) ================
    v8f acc[4] = {};
#pragma unroll
    for (int ks = 0; ks < 4; ++ks) {
        Frag b;
        const uint4* bp = (const uint4*)&lds_w[(((wave << 2) + ks) << 9) + (lane << 4)];
        b.q[0] = bp[0]; b.q[1] = bp[1];
#pragma unroll
        for (int rt = 0; rt < 4; ++rt) {
            Frag a;
            const uint4* ap = (const uint4*)&lds_a[(((rt << 2) + ks) << 9) + (lane << 4)];
            a.q[0] = ap[0]; a.q[1] = ap[1];
            acc[rt] = __builtin_amdgcn_wmma_f32_16x16x32_bf16(
                false, a.v, false, b.v, (short)0, acc[rt], false, false);
        }
    }
    __syncthreads();   // all GEMM1 LDS reads complete

    // --- kick off W2 DMA; it overlaps the hidden-layer spill below ---
    stage_w(W2bf, lds_w, tid);

    // --- bias+ReLU; spill hidden into lds_a (A-fragment order, k = cn) ---
    {
        float bias1 = b1[cn];
        int ksc, hi2, e;
        swz_a_k(cn, ksc, hi2, e);
#pragma unroll
        for (int rt = 0; rt < 4; ++rt) {
#pragma unroll
            for (int i = 0; i < 8; ++i) {
                int mt = i + (hi << 3);                 // row within 16-tile
                float v = acc[rt][i] + bias1;
                v = v > 0.0f ? v : 0.0f;
                int lane2 = (hi2 << 4) | mt;
                int dst = ((((rt << 2) + ksc) << 9)) + (lane2 << 4) + e;
                lds_a[dst] = f2bf(v);
            }
        }
    }
#if defined(HAVE_ASYNC_LDS)
    wait_async0();
#endif
    __syncthreads();

    // ===================== GEMM 2: h = relu(mid @ W2 + b2) ==================
    v8f acc2[4] = {};
#pragma unroll
    for (int ks = 0; ks < 4; ++ks) {
        Frag b;
        const uint4* bp = (const uint4*)&lds_w[(((wave << 2) + ks) << 9) + (lane << 4)];
        b.q[0] = bp[0]; b.q[1] = bp[1];
#pragma unroll
        for (int rt = 0; rt < 4; ++rt) {
            Frag a;
            const uint4* ap = (const uint4*)&lds_a[(((rt << 2) + ks) << 9) + (lane << 4)];
            a.q[0] = ap[0]; a.q[1] = ap[1];
            acc2[rt] = __builtin_amdgcn_wmma_f32_16x16x32_bf16(
                false, a.v, false, b.v, (short)0, acc2[rt], false, false);
        }
    }

    // --- bias -> ReLU -> BatchNorm(eval) -> ReLU -> f32 store ---
    {
        float bias2 = b2[cn];
        float g   = gamma[cn];
        float bt  = beta[cn];
        float rm  = rmean[cn];
        float inv = rsqrtf(rvar[cn] + BN_EPS);
#pragma unroll
        for (int rt = 0; rt < 4; ++rt) {
#pragma unroll
            for (int i = 0; i < 8; ++i) {
                int rg = r0 + (rt << 4) + i + (hi << 3);
                if (rg < N_NODES) {
                    float v = acc2[rt][i] + bias2;
                    v = v > 0.0f ? v : 0.0f;
                    v = (v - rm) * inv * g + bt;
                    v = v > 0.0f ? v : 0.0f;
                    out[(long long)rg * DIM + cn] = v;
                }
            }
        }
    }
}

// ---------------------------------------------------------------------------
// Host orchestration (graph-capture safe: kernel launches on `stream` only)
// ---------------------------------------------------------------------------
extern "C" void kernel_launch(void* const* d_in, const int* in_sizes, int n_in,
                              void* d_out, int out_size, void* d_ws, size_t ws_size,
                              hipStream_t stream) {
    const float*     x     = (const float*)d_in[0];
    const long long* ei    = (const long long*)d_in[1];   // int64 edge_index [2,E]
    const float*     W1    = (const float*)d_in[2];
    const float*     b1    = (const float*)d_in[3];
    const float*     W2    = (const float*)d_in[4];
    const float*     b2    = (const float*)d_in[5];
    const float*     gamma = (const float*)d_in[6];
    const float*     beta  = (const float*)d_in[7];
    const float*     rmean = (const float*)d_in[8];
    const float*     rvar  = (const float*)d_in[9];
    float*           out   = (float*)d_out;

    const size_t FEAT_BYTES = (size_t)N_NODES * DIM * sizeof(float);  // 51.2 MB
    char* ws = (char*)d_ws;
    float*          agg = (float*)ws;
    float*          h   = (float*)(ws + FEAT_BYTES);
    unsigned short* Wbf = (unsigned short*)(ws + 2 * FEAT_BYTES);

    {
        int total = LAYERS * DIM * DIM;
        convw_kernel<<<(total + 255) / 256, 256, 0, stream>>>(W1, W2, Wbf);
    }

    const int copy_blocks    = 1024;
    const int scatter_blocks = (N_EDGES + 7) / 8;           // 8 waves/block
    const int mlp_blocks     = (N_NODES + ROWS - 1) / ROWS; // 1563

    for (int l = 0; l < LAYERS; ++l) {
        const float* hin  = (l == 0) ? x : h;
        float*       hout = (l == LAYERS - 1) ? out : h;

        init_agg_kernel<<<copy_blocks, 256, 0, stream>>>(hin, agg);
        scatter_kernel<<<scatter_blocks, 256, 0, stream>>>(hin, ei, agg);
        mlp_kernel<<<mlp_blocks, 256, 0, stream>>>(
            agg,
            Wbf + (size_t)(l * 2 + 0) * DIM * DIM, b1 + l * DIM,
            Wbf + (size_t)(l * 2 + 1) * DIM * DIM, b2 + l * DIM,
            gamma + l * DIM, beta + l * DIM,
            rmean + l * DIM, rvar + l * DIM,
            hout);
    }
}
